// Block_1245540516098
// MI455X (gfx1250) — compile-verified, matrix operand order checked by reading, not run on hardware
//
#include <hip/hip_runtime.h>
#include <hip/hip_bf16.h>

// ---------------------------------------------------------------------------
// MI455X (gfx1250) transformer block.
// Compute-bound (~275 GFLOP vs ~80MB unique bytes -> HBM floor ~3.5us at
// 23.3 TB/s), so all GEMMs run on v_wmma_f32_16x16x32_bf16 (wave32, 16x16x32).
// Attention is streamed flash-style so the (B,H,T,T) score tensor never hits
// HBM. bf16 operands / f32 accumulation everywhere. GEMM K-loop is double-
// buffered in LDS, using GLOBAL_LOAD_ASYNC_TO_LDS_B128 (ASYNCcnt) when the
// toolchain exposes the builtin, else a manual reg-staged pipeline.
// ---------------------------------------------------------------------------

typedef __attribute__((ext_vector_type(16))) __bf16       v16bf;
typedef __attribute__((ext_vector_type(8)))  float        v8f;
typedef __attribute__((ext_vector_type(4)))  unsigned int v4u;
typedef int v4i_gcc __attribute__((vector_size(16)));      // builtin's pointee type

#define B_  4
#define T_  2048
#define D_  1024
#define H_  16
#define HS_ 64
#define M_  (B_ * T_)          // 8192 rows of activations

#if defined(__has_builtin)
#if __has_builtin(__builtin_amdgcn_global_load_async_to_lds_b128) && \
    __has_builtin(__builtin_amdgcn_s_wait_asynccnt)
#define USE_ASYNC_LDS 1
#endif
#endif
#ifndef USE_ASYNC_LDS
#define USE_ASYNC_LDS 0
#endif

#define AS1 __attribute__((address_space(1)))
#define AS3 __attribute__((address_space(3)))

#if USE_ASYNC_LDS
// Generic->AS3 via 32-bit truncation (flat LDS aperture keeps the LDS byte
// offset in addr[31:0] per the CDNA5 aperture rules); generic->AS1 is a
// value-preserving cast. Integer round-trip sidesteps clang addrspace-cast
// restrictions in HIP. Pointee type is the builtin's expected int4.
__device__ __forceinline__ AS3 v4i_gcc* to_lds(const void* p) {
    return (AS3 v4i_gcc*)(unsigned int)(unsigned long long)p;
}
__device__ __forceinline__ AS1 v4i_gcc* to_glob(const void* p) {
    return (AS1 v4i_gcc*)(unsigned long long)p;
}
#endif

__device__ __forceinline__ unsigned short f2bf(float f) {
    unsigned int u = __float_as_uint(f);
    u += 0x7FFFu + ((u >> 16) & 1u);     // round-to-nearest-even
    return (unsigned short)(u >> 16);
}

union ABFrag {                 // 16 bf16 = 8 VGPRs; two 16B LDS/global loads
    v16bf bf;
    struct { v4u lo, hi; } q;
};

// ---------------------------------------------------------------------------
// LayerNorm (f32 in) -> bf16 out.  One block per row, 256 threads, float4 IO.
// ---------------------------------------------------------------------------
__global__ __launch_bounds__(256) void ln_bf16_kernel(
    const float* __restrict__ x, const float* __restrict__ g,
    const float* __restrict__ be, unsigned short* __restrict__ out)
{
    const int row = blockIdx.x;
    const float* xr = x + (size_t)row * D_;
    float4 xv = ((const float4*)xr)[threadIdx.x];

    float s  = xv.x + xv.y + xv.z + xv.w;
    float sq = xv.x*xv.x + xv.y*xv.y + xv.z*xv.z + xv.w*xv.w;
    for (int off = 16; off >= 1; off >>= 1) {
        s  += __shfl_xor(s,  off, 32);
        sq += __shfl_xor(sq, off, 32);
    }
    __shared__ float ws[8], wq[8];
    const int wave = threadIdx.x >> 5;
    if ((threadIdx.x & 31) == 0) { ws[wave] = s; wq[wave] = sq; }
    __syncthreads();
    s = 0.f; sq = 0.f;
    for (int i = 0; i < 8; ++i) { s += ws[i]; sq += wq[i]; }

    const float mean = s * (1.0f / D_);
    const float var  = sq * (1.0f / D_) - mean * mean;
    const float rs   = rsqrtf(var + 1e-5f);

    const int c = threadIdx.x * 4;
    ushort4 ov;
    ov.x = f2bf((xv.x - mean) * rs * g[c + 0] + be[c + 0]);
    ov.y = f2bf((xv.y - mean) * rs * g[c + 1] + be[c + 1]);
    ov.z = f2bf((xv.z - mean) * rs * g[c + 2] + be[c + 2]);
    ov.w = f2bf((xv.w - mean) * rs * g[c + 3] + be[c + 3]);
    ((ushort4*)(out + (size_t)row * D_))[threadIdx.x] = ov;
}

// ---------------------------------------------------------------------------
// Weight packing: fp32 -> bf16, transposed to [N][K] (k contiguous) so the
// WMMA B-fragment reads 16B-contiguous k runs.
// ---------------------------------------------------------------------------
__global__ __launch_bounds__(256) void pack_qkv_kernel(
    const float* __restrict__ Wq, const float* __restrict__ Wk,
    const float* __restrict__ Wv, unsigned short* __restrict__ out)
{
    const int idx = blockIdx.x * 256 + threadIdx.x;    // 3 * 1024 * 1024
    const int w = idx >> 20;
    const int r = idx & 0xFFFFF;
    const int n = r >> 10;                  // h*HS + s
    const int d = r & 1023;
    const int h = n >> 6, sdim = n & 63;
    const float* W = (w == 0) ? Wq : ((w == 1) ? Wk : Wv);
    out[idx] = f2bf(W[((size_t)h * D_ + d) * HS_ + sdim]);
}

__global__ __launch_bounds__(256) void pack_t_kernel(
    const float* __restrict__ W, unsigned short* __restrict__ out,
    int K, int N)
{
    const int idx = blockIdx.x * 256 + threadIdx.x;    // N * K
    const int n = idx / K, k = idx - n * K;
    out[idx] = f2bf(W[(size_t)k * N + n]);
}

// ---------------------------------------------------------------------------
// bf16 WMMA GEMM: C[M][N] = A[M][K] * Bt[N][K]^T  (+ epilogue)
//   MODE 0: scatter-store bf16 into q/k/v [B][H][T][HS]
//   MODE 1: f32 out = C + bias[n] + res[m][n]
//   MODE 2: bf16 out = relu(C + bias[n])
// Block tile 128x128, BK=32, 8 waves (wave32) in a 4x2 grid, 32x64 per wave.
// Double-buffered LDS pipeline: tile kt+1 streams in (async-to-LDS when
// available) while tile kt feeds the WMMA pipe.
// ---------------------------------------------------------------------------
template <int MODE>
__global__ __launch_bounds__(256) void gemm_bf16_kernel(
    const unsigned short* __restrict__ A,
    const unsigned short* __restrict__ Bt,
    const float* __restrict__ bias,
    const float* __restrict__ res,
    void* __restrict__ out,
    int M, int N, int K)
{
    constexpr int LD = 40;                      // bf16 elems per LDS row (16B aligned)
    __shared__ unsigned short As[2][128 * LD];
    __shared__ unsigned short Bs[2][128 * LD];

    const int tid  = threadIdx.x;
    const int lane = tid & 31;
    const int wave = tid >> 5;
    const int wm   = wave & 3;                  // wave M coordinate (0..3)
    const int wn   = wave >> 2;                 // wave N coordinate (0..1)
    const int half = lane >> 4;                 // k-half selector
    const int l16  = lane & 15;

    const int mBase = blockIdx.y * 128;
    const int nBase = blockIdx.x * 128;

    // this thread's two 16B copy chunks of a 128x32 tile
    int rowc[2], c8c[2];
    for (int it = 0; it < 2; ++it) {
        const int chunk = tid + it * 256;       // 512 chunks of 8 elems
        rowc[it] = chunk >> 2;
        c8c[it]  = (chunk & 3) * 8;
    }

    v8f acc[2][4];
    for (int i = 0; i < 2; ++i)
        for (int j = 0; j < 4; ++j) acc[i][j] = {};

    const int kTiles = K >> 5;

    // ---- prologue: stage tile 0 into buffer 0 ----
#if USE_ASYNC_LDS
    for (int it = 0; it < 2; ++it) {
        __builtin_amdgcn_global_load_async_to_lds_b128(
            to_glob(A + (size_t)(mBase + rowc[it]) * K + c8c[it]),
            to_lds(&As[0][rowc[it] * LD + c8c[it]]), 0, 0);
        __builtin_amdgcn_global_load_async_to_lds_b128(
            to_glob(Bt + (size_t)(nBase + rowc[it]) * K + c8c[it]),
            to_lds(&Bs[0][rowc[it] * LD + c8c[it]]), 0, 0);
    }
    __builtin_amdgcn_s_wait_asynccnt(0);
    __syncthreads();
#else
    for (int it = 0; it < 2; ++it) {
        v4u av = *(const v4u*)(A  + (size_t)(mBase + rowc[it]) * K + c8c[it]);
        *(v4u*)&As[0][rowc[it] * LD + c8c[it]] = av;
        v4u bv = *(const v4u*)(Bt + (size_t)(nBase + rowc[it]) * K + c8c[it]);
        *(v4u*)&Bs[0][rowc[it] * LD + c8c[it]] = bv;
    }
    __syncthreads();
#endif

    for (int kt = 0; kt < kTiles; ++kt) {
        const int cur   = kt & 1;
        const int nxt   = cur ^ 1;
        const int kNext = (kt + 1) << 5;
        const bool more = (kt + 1 < kTiles);

#if USE_ASYNC_LDS
        if (more)                                // stream tile kt+1 (ASYNCcnt)
            for (int it = 0; it < 2; ++it) {
                __builtin_amdgcn_global_load_async_to_lds_b128(
                    to_glob(A + (size_t)(mBase + rowc[it]) * K + kNext + c8c[it]),
                    to_lds(&As[nxt][rowc[it] * LD + c8c[it]]), 0, 0);
                __builtin_amdgcn_global_load_async_to_lds_b128(
                    to_glob(Bt + (size_t)(nBase + rowc[it]) * K + kNext + c8c[it]),
                    to_lds(&Bs[nxt][rowc[it] * LD + c8c[it]]), 0, 0);
            }
#endif
        if (more) {                              // global_prefetch_b8 (L2 warm)
            __builtin_prefetch(A  + (size_t)(mBase + (tid >> 1)) * K + kNext, 0, 0);
            __builtin_prefetch(Bt + (size_t)(nBase + (tid >> 1)) * K + kNext, 0, 0);
        }

        ABFrag fa[2], fb[4];
        for (int i = 0; i < 2; ++i) {
            const int m = wm * 32 + i * 16 + l16;
            fa[i].q.lo = *(const v4u*)&As[cur][m * LD + half * 8];
            fa[i].q.hi = *(const v4u*)&As[cur][m * LD + 16 + half * 8];
        }
        for (int j = 0; j < 4; ++j) {
            const int n = wn * 64 + j * 16 + l16;
            fb[j].q.lo = *(const v4u*)&Bs[cur][n * LD + half * 8];
            fb[j].q.hi = *(const v4u*)&Bs[cur][n * LD + 16 + half * 8];
        }

#if !USE_ASYNC_LDS
        v4u anx[2] = {}, bnx[2] = {};            // reg-staged next tile: loads
        if (more)                                // issue before the WMMA burst
            for (int it = 0; it < 2; ++it) {
                anx[it] = *(const v4u*)(A  + (size_t)(mBase + rowc[it]) * K + kNext + c8c[it]);
                bnx[it] = *(const v4u*)(Bt + (size_t)(nBase + rowc[it]) * K + kNext + c8c[it]);
            }
#endif

        for (int i = 0; i < 2; ++i)
            for (int j = 0; j < 4; ++j)
                acc[i][j] = __builtin_amdgcn_wmma_f32_16x16x32_bf16(
                    false, fa[i].bf, false, fb[j].bf,
                    (short)0, acc[i][j], false, false);

#if USE_ASYNC_LDS
        __builtin_amdgcn_s_wait_asynccnt(0);
        __syncthreads();
#else
        if (more)
            for (int it = 0; it < 2; ++it) {
                *(v4u*)&As[nxt][rowc[it] * LD + c8c[it]] = anx[it];
                *(v4u*)&Bs[nxt][rowc[it] * LD + c8c[it]] = bnx[it];
            }
        __syncthreads();
#endif
    }

    for (int i = 0; i < 2; ++i)
        for (int j = 0; j < 4; ++j)
            for (int r = 0; r < 8; ++r) {
                const int m = mBase + wm * 32 + i * 16 + r + 8 * half;
                const int n = nBase + wn * 64 + j * 16 + l16;
                const float c = acc[i][j][r];
                if constexpr (MODE == 0) {
                    unsigned short* qkv = (unsigned short*)out;
                    const int w = n >> 10, n1 = n & 1023;
                    const int h = n1 >> 6, sdim = n1 & 63;
                    const int b = m >> 11, t = m & 2047;
                    qkv[(size_t)w * ((size_t)B_ * H_ * T_ * HS_) +
                        (((size_t)(b * H_ + h) * T_ + t) * HS_ + sdim)] = f2bf(c);
                } else if constexpr (MODE == 1) {
                    float* o = (float*)out;
                    o[(size_t)m * N + n] = c + bias[n] + res[(size_t)m * N + n];
                } else {
                    unsigned short* o = (unsigned short*)out;
                    const float v = c + bias[n];
                    o[(size_t)m * N + n] = f2bf(v > 0.f ? v : 0.f);
                }
            }
}

// ---------------------------------------------------------------------------
// Causal flash attention. One block per (b, h, 128-row tile of T). 8 waves;
// each wave owns 16 query rows. S = Q K^T and O += P V both use WMMA.
// Q/K fragments come straight from global (k = hs is contiguous); V is
// transposed into LDS; P round-trips through LDS to re-shape accum -> A frag.
// ---------------------------------------------------------------------------
__global__ __launch_bounds__(256) void attn_kernel(
    const unsigned short* __restrict__ q,
    const unsigned short* __restrict__ k,
    const unsigned short* __restrict__ v,
    unsigned short* __restrict__ attn)          // [B*T][D] bf16 (heads concat)
{
    constexpr int PLD = 136;                    // padded row (16B aligned)
    __shared__ unsigned short Ps[128 * PLD];    // P tile  (128 x 128)
    __shared__ unsigned short Vt[64 * PLD];     // V^T     (64  x 128)

    const int tid  = threadIdx.x;
    const int lane = tid & 31;
    const int wave = tid >> 5;
    const int half = lane >> 4;
    const int l16  = lane & 15;

    const int tb = blockIdx.x;                  // 0..15 query tile
    const int h  = blockIdx.y;
    const int b  = blockIdx.z;
    const int t0 = tb * 128;

    const size_t headOff = (size_t)(b * H_ + h) * T_ * HS_;
    const unsigned short* qh = q + headOff;
    const unsigned short* kh = k + headOff;
    const unsigned short* vh = v + headOff;

    ABFrag fq[2];                               // Q rows for this wave (HS=64 -> 2 k-steps)
    {
        const int t = t0 + wave * 16 + l16;
        for (int jf = 0; jf < 2; ++jf) {
            fq[jf].q.lo = *(const v4u*)(qh + (size_t)t * HS_ + jf * 32 + half * 8);
            fq[jf].q.hi = *(const v4u*)(qh + (size_t)t * HS_ + jf * 32 + 16 + half * 8);
        }
    }

    float row_max[8], row_sum[8];
    v8f o[4];
    for (int r = 0; r < 8; ++r) { row_max[r] = -__builtin_inff(); row_sum[r] = 0.f; }
    for (int j = 0; j < 4; ++j) o[j] = {};

    const int jmax = tb;                        // causal: kv tiles 0..tb
    for (int jb = 0; jb <= jmax; ++jb) {
        const int u0 = jb * 128;

        __syncthreads();                        // prior iter done with Vt/Ps
        for (int it = 0; it < 4; ++it) {        // Vt[s][u] = V[u0+u][s]
            const int chunk = tid + it * 256;
            const int u  = chunk >> 3;
            const int s0 = (chunk & 7) * 8;
            v4u vv = *(const v4u*)(vh + (size_t)(u0 + u) * HS_ + s0);
            unsigned short tmp[8];
            *(v4u*)tmp = vv;
            for (int e = 0; e < 8; ++e) Vt[(s0 + e) * PLD + u] = tmp[e];
        }
        __syncthreads();

        // ---- S = Q K^T : 8 tiles of 16x16, K-frags straight from global ----
        v8f sacc[8];
        for (int nt = 0; nt < 8; ++nt) sacc[nt] = {};
        for (int nt = 0; nt < 8; ++nt) {
            const int u = u0 + nt * 16 + l16;
            for (int jf = 0; jf < 2; ++jf) {
                ABFrag fk;
                fk.q.lo = *(const v4u*)(kh + (size_t)u * HS_ + jf * 32 + half * 8);
                fk.q.hi = *(const v4u*)(kh + (size_t)u * HS_ + jf * 32 + 16 + half * 8);
                sacc[nt] = __builtin_amdgcn_wmma_f32_16x16x32_bf16(
                    false, fq[jf].bf, false, fk.bf, (short)0, sacc[nt], false, false);
            }
        }

        // ---- online softmax (scale 1/sqrt(64) = 0.125, causal mask) ----
        const bool diag = (jb == jmax);
        float tmax[8];
        for (int r = 0; r < 8; ++r) tmax[r] = -__builtin_inff();
        for (int nt = 0; nt < 8; ++nt)
            for (int r = 0; r < 8; ++r) {
                float s = sacc[nt][r] * 0.125f;
                if (diag) {
                    const int t = t0 + wave * 16 + r + 8 * half;
                    const int u = u0 + nt * 16 + l16;
                    if (u > t) s = -__builtin_inff();
                }
                sacc[nt][r] = s;
                tmax[r] = fmaxf(tmax[r], s);
            }
        for (int r = 0; r < 8; ++r) {           // reduce across the 16-lane half
            float m = tmax[r];
            for (int off = 1; off < 16; off <<= 1)
                m = fmaxf(m, __shfl_xor(m, off, 32));
            const float mnew  = fmaxf(row_max[r], m);
            const float scale = __expf(row_max[r] - mnew);
            row_max[r] = mnew;
            row_sum[r] *= scale;
            for (int j = 0; j < 4; ++j) o[j][r] *= scale;
        }
        for (int nt = 0; nt < 8; ++nt)
            for (int r = 0; r < 8; ++r)
                sacc[nt][r] = __expf(sacc[nt][r] - row_max[r]);
        for (int r = 0; r < 8; ++r) {
            float s = 0.f;
            for (int nt = 0; nt < 8; ++nt) s += sacc[nt][r];
            for (int off = 1; off < 16; off <<= 1)
                s += __shfl_xor(s, off, 32);
            row_sum[r] += s;
        }
        for (int nt = 0; nt < 8; ++nt)          // P -> LDS (own 16 rows only)
            for (int r = 0; r < 8; ++r)
                Ps[(wave * 16 + r + 8 * half) * PLD + nt * 16 + l16] =
                    f2bf(sacc[nt][r]);
        __syncthreads();

        // ---- O += P V : 4 output tiles over HS, k over 128 in 4 steps ----
        for (int kk = 0; kk < 4; ++kk) {
            ABFrag fp;
            const int m = wave * 16 + l16;
            fp.q.lo = *(const v4u*)&Ps[m * PLD + kk * 32 + half * 8];
            fp.q.hi = *(const v4u*)&Ps[m * PLD + kk * 32 + 16 + half * 8];
            for (int nt = 0; nt < 4; ++nt) {
                ABFrag fv;
                const int n = nt * 16 + l16;
                fv.q.lo = *(const v4u*)&Vt[n * PLD + kk * 32 + half * 8];
                fv.q.hi = *(const v4u*)&Vt[n * PLD + kk * 32 + 16 + half * 8];
                o[nt] = __builtin_amdgcn_wmma_f32_16x16x32_bf16(
                    false, fp.bf, false, fv.bf, (short)0, o[nt], false, false);
            }
        }
    }

    for (int nt = 0; nt < 4; ++nt)              // normalize + concat heads
        for (int r = 0; r < 8; ++r) {
            const int t = t0 + wave * 16 + r + 8 * half;
            const float ov = o[nt][r] / row_sum[r];
            attn[(size_t)(b * T_ + t) * D_ + h * HS_ + nt * 16 + l16] = f2bf(ov);
        }
}

// ---------------------------------------------------------------------------
extern "C" void kernel_launch(void* const* d_in, const int* in_sizes, int n_in,
                              void* d_out, int out_size, void* d_ws, size_t ws_size,
                              hipStream_t stream) {
    const float* x   = (const float*)d_in[0];
    const float* Wq  = (const float*)d_in[1];
    const float* Wk  = (const float*)d_in[2];
    const float* Wv  = (const float*)d_in[3];
    const float* Wo  = (const float*)d_in[4];
    const float* bo  = (const float*)d_in[5];
    const float* W1  = (const float*)d_in[6];
    const float* b1  = (const float*)d_in[7];
    const float* W2  = (const float*)d_in[8];
    const float* b2  = (const float*)d_in[9];
    const float* g1  = (const float*)d_in[10];
    const float* be1 = (const float*)d_in[11];
    const float* g2  = (const float*)d_in[12];
    const float* be2 = (const float*)d_in[13];

    // ---- workspace layout (aliased; ~136 MB) ----
    char* ws = (char*)d_ws;
    unsigned short* hln  = (unsigned short*)ws;            // 16 MB (LN1, reused for LN2)
    ws += (size_t)M_ * D_ * 2;
    unsigned short* wqkv = (unsigned short*)ws;            // 6 MB
    ws += (size_t)3 * D_ * D_ * 2;
    char* region2 = ws;                                    // 64 MB: qkv+attn, later a1
    unsigned short* qkv  = (unsigned short*)region2;       // 48 MB
    unsigned short* attn = (unsigned short*)(region2 + (size_t)3 * B_ * H_ * T_ * HS_ * 2);
    unsigned short* a1   = (unsigned short*)region2;       // 64 MB (after Wo GEMM)
    ws += (size_t)M_ * 4 * D_ * 2;
    unsigned short* wo_t = (unsigned short*)ws; ws += (size_t)D_ * D_ * 2;       // 2 MB
    unsigned short* w1_t = (unsigned short*)ws; ws += (size_t)D_ * 4 * D_ * 2;   // 8 MB
    unsigned short* w2_t = (unsigned short*)ws; ws += (size_t)4 * D_ * D_ * 2;   // 8 MB
    float* x1 = (float*)ws;                                // 32 MB
    (void)ws_size; (void)in_sizes; (void)n_in; (void)out_size;

    const size_t headElems = (size_t)B_ * H_ * T_ * HS_;   // per q/k/v

    // ---- weight packing (bf16, [N][K]) ----
    pack_qkv_kernel<<<(3 * D_ * D_) / 256, 256, 0, stream>>>(Wq, Wk, Wv, wqkv);
    pack_t_kernel<<<(D_ * D_) / 256, 256, 0, stream>>>(Wo, wo_t, D_, D_);
    pack_t_kernel<<<(D_ * 4 * D_) / 256, 256, 0, stream>>>(W1, w1_t, D_, 4 * D_);
    pack_t_kernel<<<(4 * D_ * D_) / 256, 256, 0, stream>>>(W2, w2_t, 4 * D_, D_);

    // ---- attention sublayer ----
    ln_bf16_kernel<<<M_, 256, 0, stream>>>(x, g1, be1, hln);
    gemm_bf16_kernel<0><<<dim3(3 * D_ / 128, M_ / 128), 256, 0, stream>>>(
        hln, wqkv, nullptr, nullptr, qkv, M_, 3 * D_, D_);
    attn_kernel<<<dim3(T_ / 128, H_, B_), 256, 0, stream>>>(
        qkv, qkv + headElems, qkv + 2 * headElems, attn);
    gemm_bf16_kernel<1><<<dim3(D_ / 128, M_ / 128), 256, 0, stream>>>(
        attn, wo_t, bo, x, x1, M_, D_, D_);

    // ---- feed-forward sublayer ----
    ln_bf16_kernel<<<M_, 256, 0, stream>>>(x1, g2, be2, hln);
    gemm_bf16_kernel<2><<<dim3(4 * D_ / 128, M_ / 128), 256, 0, stream>>>(
        hln, w1_t, b1, nullptr, a1, M_, 4 * D_, D_);
    gemm_bf16_kernel<1><<<dim3(D_ / 128, M_ / 128), 256, 0, stream>>>(
        a1, w2_t, b2, x1, (float*)d_out, M_, D_, 4 * D_);
}
